// MetaLSTMCell_35691178230152
// MI455X (gfx1250) — matrix operand order repaired and checked by reading, not compiled, not run on hardware
//
#include <hip/hip_runtime.h>

typedef __bf16 bf16;
typedef __attribute__((ext_vector_type(16))) __bf16 v16bf;
typedef __attribute__((ext_vector_type(8)))  __bf16 v8bf;
typedef __attribute__((ext_vector_type(8)))  float  v8f;
typedef __attribute__((__vector_size__(16))) int    v4i_vs;   // 16B chunk type for async-LDS builtin

#define DEVINL static __device__ __forceinline__
#define AS1 __attribute__((address_space(1)))
#define AS3 __attribute__((address_space(3)))

// Async global->LDS path (CDNA5 GLOBAL_LOAD_ASYNC_TO_LDS_*, ASYNCcnt-tracked).
#if defined(__AMDGCN__) && __has_builtin(__builtin_amdgcn_global_load_async_to_lds_b128)
#define HAS_ASYNC_LDS 1
#else
#define HAS_ASYNC_LDS 0
#endif

constexpr int Bsz  = 4096;  // batch
constexpr int INsz = 1024;  // input size
constexpr int Hsz  = 1024;  // main hidden
constexpr int HHsz = 256;   // hyper hidden
constexpr int Esz  = 64;    // hyper embedding
constexpr int BK   = 32;    // k-tile (wmma bf16 K)
constexpr int PADK = 40;    // LDS row stride in elements (80B: 16B-aligned, bank-skewed)

// ------------------------------------------------------------------
// fp32 -> bf16 bulk conversion (all tensor sizes here are multiples of 8)
// ------------------------------------------------------------------
__global__ __launch_bounds__(256) void cvt_f32_to_bf16(const float* __restrict__ src,
                                                       bf16* __restrict__ dst, int n8) {
  int i = blockIdx.x * blockDim.x + threadIdx.x;
  if (i >= n8) return;
  const float* s = src + (size_t)i * 8;
  float4 a = *(const float4*)(s);
  float4 b = *(const float4*)(s + 4);
  v8bf o;
  o[0] = (bf16)a.x; o[1] = (bf16)a.y; o[2] = (bf16)a.z; o[3] = (bf16)a.w;
  o[4] = (bf16)b.x; o[5] = (bf16)b.y; o[6] = (bf16)b.z; o[7] = (bf16)b.w;
  *(v8bf*)(dst + (size_t)i * 8) = o;
}

// ------------------------------------------------------------------
// WMMA fragment loads from an LDS k-tile (row stride PADK elements)
// A 16x32 bf16 (ISA 7.12.2): lane&15 = row M; lane>>4 selects K-half.
//   v0..3 = K[kh..kh+7], v4..7 = K[kh+16..kh+23], kh = 8*(lane>>4)
// ------------------------------------------------------------------
DEVINL v16bf load_frag_a(const bf16* s, int lane) {
  int r  = lane & 15;
  int kh = (lane >> 4) << 3;
  const bf16* p = s + r * PADK + kh;
  v8bf lo = *(const v8bf*)(p);
  v8bf hi = *(const v8bf*)(p + 16);
  return __builtin_shufflevector(lo, hi, 0,1,2,3,4,5,6,7,8,9,10,11,12,13,14,15);
}
// B 32x16 bf16: lane&15 = column N; v0..3 = K[kh..kh+7], v4..7 = K[kh+16..kh+23], kh = 8*(lane>>4)
DEVINL v16bf load_frag_b(const bf16* s, int lane) {
  int c  = lane & 15;
  int kb = (lane >> 4) << 3;
  const bf16* p = s + c * PADK + kb;
  v8bf lo = *(const v8bf*)(p);
  v8bf hi = *(const v8bf*)(p + 16);
  return __builtin_shufflevector(lo, hi, 0,1,2,3,4,5,6,7,8,9,10,11,12,13,14,15);
}

// 16-byte tile-chunk mover: async DMA to LDS if available, else via VGPRs.
DEVINL void copy16_to_lds(bf16* dst, const bf16* src) {
#if HAS_ASYNC_LDS
  __builtin_amdgcn_global_load_async_to_lds_b128((AS1 v4i_vs*)src, (AS3 v4i_vs*)dst, 0, 0);
#else
  *(v8bf*)dst = *(const v8bf*)src;
#endif
}

DEVINL void pipeline_wait_barrier() {
#if HAS_ASYNC_LDS
#if __has_builtin(__builtin_amdgcn_s_wait_asynccnt)
  __builtin_amdgcn_s_wait_asynccnt(0);
#else
  asm volatile("s_wait_asynccnt 0x0" ::: "memory");
#endif
#endif
  __syncthreads();
}

// Cooperative global -> LDS tile: R rows x 32 k-cols, 256 threads, 16B chunks.
template<int R>
DEVINL void issue_tile(bf16* __restrict__ dst, const bf16* __restrict__ src, int ld) {
  constexpr int per = R * BK / 256;   // 16 (R=128) or 8 (R=64) elements per thread
  int t = threadIdx.x;
  #pragma unroll
  for (int j = 0; j < per; j += 8) {
    int e   = t * per + j;
    int row = e >> 5;
    int col = e & 31;
    copy16_to_lds(dst + row * PADK + col, src + (size_t)row * ld + col);
  }
}

// ------------------------------------------------------------------
// Generic bf16 WMMA GEMM:  D[M,N] = sum_s  A_s[M,K_s] * B_s[N,K_s]^T
// MODE 0: store fp32          MODE 1: store bf16
// MODE 2: D = D * (Z @ W1^T)                     , store fp32   (K_z = E = 64)
// MODE 3: D = prev + D * (Z @ W1^T) + (Z @ W2^T) , store fp32
// Block: 256 threads = 8 waves; wave grid WM x WN; wave tile (BM/WM) x (BN/WN).
// Main K-loop: double-buffered LDS + async-to-LDS copies, 1 barrier per k-step.
// ------------------------------------------------------------------
template<int BM, int BN, int WM, int WN, int MODE>
__global__ __launch_bounds__(256) void gemm_bf16_kernel(
    const bf16* __restrict__ A0, int lda0, const bf16* __restrict__ B0, int ldb0, int K0,
    const bf16* __restrict__ A1, int lda1, const bf16* __restrict__ B1, int ldb1, int K1,
    const bf16* __restrict__ A2, int lda2, const bf16* __restrict__ B2, int ldb2, int K2,
    int Nld, float* __restrict__ outF, bf16* __restrict__ outB,
    const float* __restrict__ prev,
    const bf16* __restrict__ Z, const bf16* __restrict__ W1, const bf16* __restrict__ W2)
{
  constexpr int MT = BM / WM / 16;
  constexpr int NT = BN / WN / 16;

  __shared__ bf16 sA[2][BM * PADK];
  __shared__ bf16 sB[2][BN * PADK];

  const int bm   = blockIdx.y * BM;
  const int bn   = blockIdx.x * BN;
  const int lane = threadIdx.x & 31;
  const int wave = threadIdx.x >> 5;
  const int wm   = wave % WM;
  const int wn   = wave / WM;
  const int rowA0 = wm * MT * 16;
  const int colB0 = wn * NT * 16;

  v8f acc[MT][NT];
  #pragma unroll
  for (int i = 0; i < MT; ++i)
    #pragma unroll
    for (int j = 0; j < NT; ++j)
      #pragma unroll
      for (int v = 0; v < 8; ++v) acc[i][j][v] = 0.0f;

  // one k-step of WMMA from LDS buffer `cur`
  auto compute = [&](int cur, v8f (&C)[MT][NT]) {
    v16bf af[MT], bfr[NT];
    #pragma unroll
    for (int i = 0; i < MT; ++i) af[i]  = load_frag_a(&sA[cur][(rowA0 + i * 16) * PADK], lane);
    #pragma unroll
    for (int j = 0; j < NT; ++j) bfr[j] = load_frag_b(&sB[cur][(colB0 + j * 16) * PADK], lane);
    #pragma unroll
    for (int i = 0; i < MT; ++i)
      #pragma unroll
      for (int j = 0; j < NT; ++j)
        C[i][j] = __builtin_amdgcn_wmma_f32_16x16x32_bf16(
            false, af[i], false, bfr[j], (short)0, C[i][j], false, false);
  };

  // double-buffered async pipeline over one operand pair
  auto mm_async = [&](const bf16* __restrict__ A, int lda,
                      const bf16* __restrict__ Bp, int ldb, int K, v8f (&C)[MT][NT]) {
    const int nk = K / BK;
    __syncthreads();                       // smem free from prior use
    issue_tile<BM>(sA[0], A, lda);
    issue_tile<BN>(sB[0], Bp, ldb);
    pipeline_wait_barrier();
    for (int kt = 0; kt < nk; ++kt) {
      const int cur = kt & 1;
      if (kt + 1 < nk) {                   // prefetch next tile into other buffer
        issue_tile<BM>(sA[cur ^ 1], A + (kt + 1) * BK, lda);
        issue_tile<BN>(sB[cur ^ 1], Bp + (kt + 1) * BK, ldb);
      }
      compute(cur, C);
      pipeline_wait_barrier();
    }
  };

  // simple synchronous path (used for the tiny K=64 epilogue GEMMs)
  auto mm_sync = [&](const bf16* __restrict__ A, int lda,
                     const bf16* __restrict__ Bp, int ldb, int K, v8f (&C)[MT][NT]) {
    for (int k = 0; k < K; k += BK) {
      __syncthreads();
      issue_tile<BM>(sA[0], A + k, lda);
      issue_tile<BN>(sB[0], Bp + k, ldb);
      pipeline_wait_barrier();
      compute(0, C);
    }
  };

  if (K0) mm_async(A0 + (size_t)bm * lda0, lda0, B0 + (size_t)bn * ldb0, ldb0, K0, acc);
  if (K1) mm_async(A1 + (size_t)bm * lda1, lda1, B1 + (size_t)bn * ldb1, ldb1, K1, acc);
  if (K2) mm_async(A2 + (size_t)bm * lda2, lda2, B2 + (size_t)bn * ldb2, ldb2, K2, acc);

  // C/D layout (ISA 7.12.2): N = lane&15, M = 8*(lane>>4) + vgpr
  const int rOff = (lane >> 4) << 3;
  const int cIn  = lane & 15;

  if constexpr (MODE == 2 || MODE == 3) {
    v8f zacc[MT][NT];
    #pragma unroll
    for (int i = 0; i < MT; ++i)
      #pragma unroll
      for (int j = 0; j < NT; ++j)
        #pragma unroll
        for (int v = 0; v < 8; ++v) zacc[i][j][v] = 0.0f;

    mm_sync(Z + (size_t)bm * Esz, Esz, W1 + (size_t)bn * Esz, Esz, Esz, zacc);

    if constexpr (MODE == 2) {
      #pragma unroll
      for (int i = 0; i < MT; ++i)
        #pragma unroll
        for (int j = 0; j < NT; ++j)
          #pragma unroll
          for (int v = 0; v < 8; ++v) acc[i][j][v] *= zacc[i][j][v];
    } else {
      #pragma unroll
      for (int i = 0; i < MT; ++i)
        #pragma unroll
        for (int j = 0; j < NT; ++j) {
          int r0 = bm + rowA0 + i * 16 + rOff;
          int c  = bn + colB0 + j * 16 + cIn;
          #pragma unroll
          for (int v = 0; v < 8; ++v)
            acc[i][j][v] = prev[(size_t)(r0 + v) * Nld + c] + acc[i][j][v] * zacc[i][j][v];
        }
      #pragma unroll
      for (int i = 0; i < MT; ++i)
        #pragma unroll
        for (int j = 0; j < NT; ++j)
          #pragma unroll
          for (int v = 0; v < 8; ++v) zacc[i][j][v] = 0.0f;
      mm_sync(Z + (size_t)bm * Esz, Esz, W2 + (size_t)bn * Esz, Esz, Esz, zacc);
      #pragma unroll
      for (int i = 0; i < MT; ++i)
        #pragma unroll
        for (int j = 0; j < NT; ++j)
          #pragma unroll
          for (int v = 0; v < 8; ++v) acc[i][j][v] += zacc[i][j][v];
    }
  }

  #pragma unroll
  for (int i = 0; i < MT; ++i)
    #pragma unroll
    for (int j = 0; j < NT; ++j) {
      int r0 = bm + rowA0 + i * 16 + rOff;
      int c  = bn + colB0 + j * 16 + cIn;
      if constexpr (MODE == 1) {
        #pragma unroll
        for (int v = 0; v < 8; ++v)
          outB[(size_t)(r0 + v) * Nld + c] = (bf16)acc[i][j][v];
      } else {
        #pragma unroll
        for (int v = 0; v < 8; ++v)
          outF[(size_t)(r0 + v) * Nld + c] = acc[i][j][v];
      }
    }
}

// ------------------------------------------------------------------
// Gate nonlinearities
// ------------------------------------------------------------------
DEVINL float sigmoidf(float x) { return 1.0f / (1.0f + __expf(-x)); }

__global__ __launch_bounds__(256) void meta_gates_kernel(
    const float* __restrict__ pre, const float* __restrict__ bias,
    const float* __restrict__ c_in,
    float* __restrict__ h_out, float* __restrict__ c_out,
    bf16* __restrict__ h_bf)
{
  int idx = blockIdx.x * blockDim.x + threadIdx.x;   // over B*HH
  int b = idx >> 8;            // / HHsz
  int j = idx & (HHsz - 1);
  const float* row = pre + (size_t)b * (4 * HHsz);
  float pi = row[j]             + bias[j];
  float pf = row[HHsz + j]      + bias[HHsz + j];
  float pg = row[2 * HHsz + j]  + bias[2 * HHsz + j];
  float po = row[3 * HHsz + j]  + bias[3 * HHsz + j];
  float i = sigmoidf(pi), f = sigmoidf(pf), o = sigmoidf(po);
  float g = tanhf(pg);
  float c = f * c_in[idx] + i * g;
  float h = o * tanhf(c);
  c_out[idx] = c;
  h_out[idx] = h;
  h_bf[idx]  = (bf16)h;
}

__global__ __launch_bounds__(256) void main_gates_kernel(
    const float* __restrict__ pre, const float* __restrict__ c_in,
    float* __restrict__ h_out, float* __restrict__ c_out)
{
  int idx = blockIdx.x * blockDim.x + threadIdx.x;   // over B*H
  int b = idx >> 10;           // / Hsz
  int j = idx & (Hsz - 1);
  const float* row = pre + (size_t)b * (4 * Hsz);
  float pi = row[j];
  float pf = row[Hsz + j];
  float pg = row[2 * Hsz + j];
  float po = row[3 * Hsz + j];
  float i = sigmoidf(pi), f = sigmoidf(pf), o = sigmoidf(po);
  float g = tanhf(pg);
  float c = f * c_in[idx] + i * g;
  float h = o * tanhf(c);
  c_out[idx] = c;
  h_out[idx] = h;
}

// ------------------------------------------------------------------
// Host-side orchestration
// ------------------------------------------------------------------
extern "C" void kernel_launch(void* const* d_in, const int* in_sizes, int n_in,
                              void* d_out, int out_size, void* d_ws, size_t ws_size,
                              hipStream_t stream) {
  (void)in_sizes; (void)n_in; (void)out_size; (void)ws_size;

  const float* x       = (const float*)d_in[0];
  const float* main_h  = (const float*)d_in[1];
  const float* main_c  = (const float*)d_in[2];
  const float* meta_h  = (const float*)d_in[3];
  const float* meta_c  = (const float*)d_in[4];
  const float* W_iH    = (const float*)d_in[5];   // [4H, IN]
  const float* W_HH    = (const float*)d_in[6];   // [4H, H]
  const float* W_ih    = (const float*)d_in[7];   // [4HH, IN+H]
  const float* W_hh    = (const float*)d_in[8];   // [4HH, HH]
  const float* W_hz    = (const float*)d_in[9];   // [E, HH]
  const float* W_dziH  = (const float*)d_in[10];  // [4H, E]
  const float* W_dzHH  = (const float*)d_in[11];  // [4H, E]
  const float* W_bzH   = (const float*)d_in[12];  // [4H, E]
  const float* b_hyper = (const float*)d_in[13];  // [4HH]

  float* out_main_h = (float*)d_out;
  float* out_main_c = out_main_h + (size_t)Bsz * Hsz;
  float* out_meta_h = out_main_c + (size_t)Bsz * Hsz;
  float* out_meta_c = out_meta_h + (size_t)Bsz * HHsz;

  // ---- workspace carve-up (256B-aligned) ----
  char* ws = (char*)d_ws;
  size_t off = 0;
  auto carve = [&](size_t bytes) -> char* {
    char* p = ws + off;
    off += (bytes + 255) & ~(size_t)255;
    return p;
  };
  bf16* x_bf    = (bf16*)carve((size_t)Bsz * INsz * 2);
  bf16* h_bf    = (bf16*)carve((size_t)Bsz * Hsz * 2);
  bf16* mh_bf   = (bf16*)carve((size_t)Bsz * HHsz * 2);
  bf16* Wih_bf  = (bf16*)carve((size_t)4 * HHsz * (INsz + Hsz) * 2);
  bf16* Whh_bf  = (bf16*)carve((size_t)4 * HHsz * HHsz * 2);
  bf16* WiH_bf  = (bf16*)carve((size_t)4 * Hsz * INsz * 2);
  bf16* WHH_bf  = (bf16*)carve((size_t)4 * Hsz * Hsz * 2);
  bf16* Whz_bf  = (bf16*)carve((size_t)Esz * HHsz * 2);
  bf16* Wdzi_bf = (bf16*)carve((size_t)4 * Hsz * Esz * 2);
  bf16* Wdzh_bf = (bf16*)carve((size_t)4 * Hsz * Esz * 2);
  bf16* Wbz_bf  = (bf16*)carve((size_t)4 * Hsz * Esz * 2);
  bf16* mhn_bf  = (bf16*)carve((size_t)Bsz * HHsz * 2);   // meta_h_new (bf16)
  bf16* z_bf    = (bf16*)carve((size_t)Bsz * Esz * 2);    // z (bf16)
  float* meta_pre = (float*)carve((size_t)Bsz * 4 * HHsz * 4);
  float* pre      = (float*)carve((size_t)Bsz * 4 * Hsz * 4);

  auto cvt = [&](const float* s, bf16* d, size_t n) {
    int n8 = (int)(n / 8);
    cvt_f32_to_bf16<<<dim3((n8 + 255) / 256), dim3(256), 0, stream>>>(s, d, n8);
  };

  // ---- 1) fp32 -> bf16 conversions ----
  cvt(x,      x_bf,    (size_t)Bsz * INsz);
  cvt(main_h, h_bf,    (size_t)Bsz * Hsz);
  cvt(meta_h, mh_bf,   (size_t)Bsz * HHsz);
  cvt(W_ih,   Wih_bf,  (size_t)4 * HHsz * (INsz + Hsz));
  cvt(W_hh,   Whh_bf,  (size_t)4 * HHsz * HHsz);
  cvt(W_iH,   WiH_bf,  (size_t)4 * Hsz * INsz);
  cvt(W_HH,   WHH_bf,  (size_t)4 * Hsz * Hsz);
  cvt(W_hz,   Whz_bf,  (size_t)Esz * HHsz);
  cvt(W_dziH, Wdzi_bf, (size_t)4 * Hsz * Esz);
  cvt(W_dzHH, Wdzh_bf, (size_t)4 * Hsz * Esz);
  cvt(W_bzH,  Wbz_bf,  (size_t)4 * Hsz * Esz);

  // ---- 2) meta_pre = [x|main_h|meta_h] @ [Wih_x|Wih_h|Whh]^T  (bias in gates) ----
  gemm_bf16_kernel<128, 128, 4, 2, 0>
      <<<dim3((4 * HHsz) / 128, Bsz / 128), dim3(256), 0, stream>>>(
      x_bf,  INsz, Wih_bf,        INsz + Hsz, INsz,
      h_bf,  Hsz,  Wih_bf + INsz, INsz + Hsz, Hsz,
      mh_bf, HHsz, Whh_bf,        HHsz,       HHsz,
      4 * HHsz, meta_pre, nullptr, nullptr, nullptr, nullptr, nullptr);

  // ---- 3) meta gates -> meta_h_new / meta_c_new (+ bf16 copy) ----
  meta_gates_kernel<<<dim3((Bsz * HHsz) / 256), dim3(256), 0, stream>>>(
      meta_pre, b_hyper, meta_c, out_meta_h, out_meta_c, mhn_bf);

  // ---- 4) z = meta_h_new @ W_hz^T  (bf16 out) ----
  gemm_bf16_kernel<128, 64, 8, 1, 1>
      <<<dim3(Esz / 64, Bsz / 128), dim3(256), 0, stream>>>(
      mhn_bf, HHsz, Whz_bf, HHsz, HHsz,
      nullptr, 0, nullptr, 0, 0,
      nullptr, 0, nullptr, 0, 0,
      Esz, nullptr, z_bf, nullptr, nullptr, nullptr, nullptr);

  // ---- 5) pre = (z @ W_dziH^T) * (x @ W_iH^T) ----
  gemm_bf16_kernel<128, 128, 4, 2, 2>
      <<<dim3((4 * Hsz) / 128, Bsz / 128), dim3(256), 0, stream>>>(
      x_bf, INsz, WiH_bf, INsz, INsz,
      nullptr, 0, nullptr, 0, 0,
      nullptr, 0, nullptr, 0, 0,
      4 * Hsz, pre, nullptr, nullptr, z_bf, Wdzi_bf, nullptr);

  // ---- 6) pre += (z @ W_dzHH^T) * (main_h @ W_HH^T) + (z @ W_bzH^T) ----
  gemm_bf16_kernel<128, 128, 4, 2, 3>
      <<<dim3((4 * Hsz) / 128, Bsz / 128), dim3(256), 0, stream>>>(
      h_bf, Hsz, WHH_bf, Hsz, Hsz,
      nullptr, 0, nullptr, 0, 0,
      nullptr, 0, nullptr, 0, 0,
      4 * Hsz, pre, nullptr, pre, z_bf, Wdzh_bf, Wbz_bf);

  // ---- 7) main gates -> main_h_new / main_c_new ----
  main_gates_kernel<<<dim3((Bsz * Hsz) / 256), dim3(256), 0, stream>>>(
      pre, main_c, out_main_h, out_main_c);
}
